// SPVCNN_25039659336250
// MI455X (gfx1250) — matrix operand order repaired and checked by reading, not compiled, not run on hardware
//
#include <hip/hip_runtime.h>

// CDNA5 / gfx1250: wave32, WMMA 16x16x32 f16 -> f32 accumulate.
// Activations live in voxel-major [N][C] f16 so B fragments are single
// contiguous 32B vector loads; weights are pre-packed into A-fragment order.
// Conv kernel is 2x2 register-blocked: 4 WMMAs per 4 fragment loads.
typedef __attribute__((ext_vector_type(16))) _Float16 v16h;
typedef __attribute__((ext_vector_type(8)))  _Float16 v8h;
typedef __attribute__((ext_vector_type(8)))  float    v8f;

#define WAVES_PER_BLOCK 8
#define BLOCK_THREADS (WAVES_PER_BLOCK * 32)

// ---------------------------------------------------------------------------
// One-time weight pack: W f32 [Cout][Cin][ks^3] -> A-fragment-ordered f16.
// Wp layout: [(mTile*ks3 + t)*KC + cc][lane][16 f16]  (KC = Cin/32)
// contiguous k order: element s of lane half h holds k = 16h + s (matches B loads).
// ---------------------------------------------------------------------------
__global__ void pack_w_f16(const float* __restrict__ W, _Float16* __restrict__ Wp,
                           int Cout, int Cin, int ks3)
{
    const int KC = Cin >> 5;
    const int total = (Cout >> 4) * ks3 * KC * 32;
    const int tid = blockIdx.x * blockDim.x + threadIdx.x;
    if (tid >= total) return;
    const int lane = tid & 31;
    int rest = tid >> 5;
    const int cc = rest % KC;  rest /= KC;
    const int t  = rest % ks3;
    const int mTile = rest / ks3;
    const int hi = (lane >> 4) & 1, l15 = lane & 15;
    const int mRow = mTile * 16 + l15;
    const float* Wb = W + ((size_t)mRow * Cin + (cc << 5)) * ks3 + t;
    v16h q;
    #pragma unroll
    for (int r = 0; r < 8; ++r) {
        const int ka = (hi << 4) + (r << 1);
        q[2 * r]     = (_Float16)Wb[(size_t)ka * ks3];
        q[2 * r + 1] = (_Float16)Wb[(size_t)(ka + 1) * ks3];
    }
    *(v16h*)(Wp + (size_t)tid * 16) = q;
}

// [C][N] f32 -> [N][C] f16 (for the network input).
__global__ void cn_to_nc_f16(const float* __restrict__ Xcn, _Float16* __restrict__ Xnc,
                             int C, int N)
{
    const int tid = blockIdx.x * blockDim.x + threadIdx.x;
    if (tid >= C * N) return;
    const int c = tid % C;
    const int n = tid / C;
    Xnc[tid] = (_Float16)Xcn[(size_t)c * N + n];
}

__global__ void zero_f16(_Float16* __restrict__ p, int n)
{
    const int tid = blockIdx.x * blockDim.x + threadIdx.x;
    if (tid < n) p[tid] = (_Float16)0.0f;
}

// ---------------------------------------------------------------------------
// Fused Conv3D + BN (+residual add) (+ReLU) (+sparse mask), implicit GEMM.
//   X:  f16 [Din^3][XCs] (channels XcOff .. XcOff+Cin-1 used)
//   Wp: packed A fragments; Zp: zeroed page for out-of-bounds taps
//   Y:  f16 [Dout^3][YCs] at channel offset YcOff   (or Y32: f32 [Cout][Dout^3])
// 2x2 blocked: per K-chunk -> 2 A loads + 2 B loads feeding 4 WMMAs.
// ---------------------------------------------------------------------------
__global__ __launch_bounds__(BLOCK_THREADS)
void conv3d_nc_wmma(const _Float16* __restrict__ X, int XCs, int XcOff,
                    const _Float16* __restrict__ Wp,
                    const _Float16* __restrict__ Zp,
                    const float* __restrict__ bg, const float* __restrict__ bb,
                    const float* __restrict__ bm, const float* __restrict__ bv,
                    const _Float16* __restrict__ addSrc, int ACs,
                    const float* __restrict__ mask,
                    _Float16* __restrict__ Y, int YCs, int YcOff,
                    float* __restrict__ Y32,
                    int Cin, int Cout, int Din, int Dout,
                    int ks, int stride, int pad, int relu)
{
    const int lane = threadIdx.x & 31;
    const int wave = threadIdx.x >> 5;
    const int Nout = Dout * Dout * Dout;
    const int MT2  = Cout >> 5;              // pairs of 16-row tiles
    const int NT2  = Nout >> 5;              // pairs of 16-voxel tiles
    const int total = MT2 * NT2;
    const int tile  = blockIdx.x * WAVES_PER_BLOCK + wave;   // wave-uniform
    if (tile >= total) return;                                // uniform exit (EXEC all-ones for WMMA)

    const int mTile = (tile % MT2) * 2;
    const int nTile = (tile / MT2) * 2;
    const int ks3 = ks * ks * ks;
    const int KC  = Cin >> 5;
    const int hi  = (lane >> 4) & 1;
    const int l15 = lane & 15;

    // Two output voxels per lane (adjacent 16-voxel tiles).
    const int n0 = nTile * 16 + l15;
    const int n1 = n0 + 16;
    const int xo0 = n0 % Dout, yo0 = (n0 / Dout) % Dout, zo0 = n0 / (Dout * Dout);
    const int xo1 = n1 % Dout, yo1 = (n1 / Dout) % Dout, zo1 = n1 / (Dout * Dout);

    const v16h* __restrict__ Ap = (const v16h*)Wp;
    const size_t mstride = (size_t)ks3 * KC * 32;      // v16h units per mTile block
    const _Float16* __restrict__ Zv = Zp + (hi << 4);

    v8f a00 = {}, a01 = {}, a10 = {}, a11 = {};
    for (int t = 0; t < ks3; ++t) {
        const int tx = t % ks, ty = (t / ks) % ks, tz = t / (ks * ks);

        const int xi0 = xo0 * stride + tx - pad;
        const int yi0 = yo0 * stride + ty - pad;
        const int zi0 = zo0 * stride + tz - pad;
        const bool inb0 = (unsigned)xi0 < (unsigned)Din &&
                          (unsigned)yi0 < (unsigned)Din &&
                          (unsigned)zi0 < (unsigned)Din;
        const _Float16* Xv0 = inb0
            ? X + (size_t)((zi0 * Din + yi0) * Din + xi0) * XCs + XcOff + (hi << 4)
            : Zv;

        const int xi1 = xo1 * stride + tx - pad;
        const int yi1 = yo1 * stride + ty - pad;
        const int zi1 = zo1 * stride + tz - pad;
        const bool inb1 = (unsigned)xi1 < (unsigned)Din &&
                          (unsigned)yi1 < (unsigned)Din &&
                          (unsigned)zi1 < (unsigned)Din;
        const _Float16* Xv1 = inb1
            ? X + (size_t)((zi1 * Din + yi1) * Din + xi1) * XCs + XcOff + (hi << 4)
            : Zv;

        const size_t ab = (size_t)(mTile * ks3 + t) * KC * 32 + lane;

        for (int cc = 0; cc < KC; ++cc) {
            const v16h wa0 = Ap[ab + (size_t)cc * 32];
            const v16h wa1 = Ap[ab + mstride + (size_t)cc * 32];
            const v16h xb0 = *(const v16h*)(Xv0 + (cc << 5));
            const v16h xb1 = *(const v16h*)(Xv1 + (cc << 5));
            a00 = __builtin_amdgcn_wmma_f32_16x16x32_f16(false, wa0, false, xb0, (short)0, a00, false, false);
            a10 = __builtin_amdgcn_wmma_f32_16x16x32_f16(false, wa1, false, xb0, (short)0, a10, false, false);
            a01 = __builtin_amdgcn_wmma_f32_16x16x32_f16(false, wa0, false, xb1, (short)0, a01, false, false);
            a11 = __builtin_amdgcn_wmma_f32_16x16x32_f16(false, wa1, false, xb1, (short)0, a11, false, false);
        }
    }

    // Epilogue: BN affine + optional residual + ReLU + sparse mask.
    const int m0 = mTile * 16 + (hi << 3);   // first of 8 consecutive C/D rows (i=0)
    #pragma unroll
    for (int j = 0; j < 2; ++j) {
        const int n = n0 + j * 16;
        const v8f accj0 = j ? a01 : a00;
        const v8f accj1 = j ? a11 : a10;
        const float mk = mask ? mask[n] : 1.0f;
        #pragma unroll
        for (int i = 0; i < 2; ++i) {
            const int mb = m0 + i * 16;
            const v8f acc = i ? accj1 : accj0;
            v8h ad = {};
            if (addSrc) ad = *(const v8h*)(addSrc + (size_t)n * ACs + mb);
            float out[8];
            #pragma unroll
            for (int r = 0; r < 8; ++r) {
                const int m = mb + r;
                const float sc = bg[m] * rsqrtf(bv[m] + 1e-5f);
                float val = acc[r] * sc + (bb[m] - bm[m] * sc);
                if (addSrc) val += (float)ad[r];
                if (relu)   val = fmaxf(val, 0.0f);
                out[r] = val * mk;
            }
            if (Y32) {
                #pragma unroll
                for (int r = 0; r < 8; ++r)
                    Y32[(size_t)(mb + r) * Nout + n] = out[r];
            } else {
                v8h o;
                #pragma unroll
                for (int r = 0; r < 8; ++r) o[r] = (_Float16)out[r];
                *(v8h*)(Y + (size_t)n * YCs + YcOff + mb) = o;
            }
        }
    }
}

// ---------------------------------------------------------------------------
// Transposed conv ks=2 stride=2 (+BN+ReLU+mask).  Grid iterates taps so all 16
// columns of a WMMA tile share one tap -> dense GEMM per tap.
//   Xc: f16 [Dc^3][Cin]   Wp: packed (ks3=8)   Y: f16 [(2Dc)^3][YCs]+YcOff
// ---------------------------------------------------------------------------
__global__ __launch_bounds__(BLOCK_THREADS)
void deconv3d_nc_wmma(const _Float16* __restrict__ X,
                      const _Float16* __restrict__ Wp,
                      const float* __restrict__ bg, const float* __restrict__ bb,
                      const float* __restrict__ bm, const float* __restrict__ bv,
                      const float* __restrict__ mask,
                      _Float16* __restrict__ Y, int YCs, int YcOff,
                      int Cin, int Cout, int Dc)
{
    const int lane = threadIdx.x & 31;
    const int wave = threadIdx.x >> 5;
    const int MT = Cout >> 4;
    const int Nc = Dc * Dc * Dc;
    const int PT = Nc >> 4;
    const int total = MT * PT * 8;
    const int tile  = blockIdx.x * WAVES_PER_BLOCK + wave;
    if (tile >= total) return;

    const int mTile = tile % MT;
    const int pTile = (tile / MT) % PT;
    const int tap   = tile / (MT * PT);
    const int hi  = (lane >> 4) & 1;
    const int l15 = lane & 15;
    const int KC  = Cin >> 5;

    const int p  = pTile * 16 + l15;                 // parent (coarse) voxel
    const int px = p % Dc, py = (p / Dc) % Dc, pz = p / (Dc * Dc);
    const int Df = Dc * 2;
    const int fx = 2 * px + (tap & 1);
    const int fy = 2 * py + ((tap >> 1) & 1);
    const int fz = 2 * pz + ((tap >> 2) & 1);
    const int nf = (fz * Df + fy) * Df + fx;         // fine output voxel

    const v16h* __restrict__ Ap = (const v16h*)Wp;
    const size_t abase = (size_t)(mTile * 8 + tap) * KC * 32 + lane;
    const _Float16* Xv = X + (size_t)p * Cin + (hi << 4);

    v8f acc = {};
    for (int cc = 0; cc < KC; ++cc) {
        const v16h a = Ap[abase + (size_t)cc * 32];
        const v16h b = *(const v16h*)(Xv + (cc << 5));
        acc = __builtin_amdgcn_wmma_f32_16x16x32_f16(
                  false, a, false, b, (short)0, acc, false, false);
    }

    const float mk = mask[nf];
    const int m0 = mTile * 16 + (hi << 3);
    v8h o;
    #pragma unroll
    for (int r = 0; r < 8; ++r) {
        const int m = m0 + r;
        const float sc = bg[m] * rsqrtf(bv[m] + 1e-5f);
        float val = acc[r] * sc + (bb[m] - bm[m] * sc);
        o[r] = (_Float16)(fmaxf(val, 0.0f) * mk);
    }
    *(v8h*)(Y + (size_t)nf * YCs + YcOff + m0) = o;
}

// ---------------------------------------------------------------------------

extern "C" void kernel_launch(void* const* d_in, const int* in_sizes, int n_in,
                              void* d_out, int out_size, void* d_ws, size_t ws_size,
                              hipStream_t stream)
{
    (void)in_sizes; (void)n_in; (void)out_size; (void)ws_size;

    struct BN { const float *g, *b, *m, *v; };
    struct CB { const float* w; BN bn; };
    struct RB { const float* w1; BN bn1; const float* w2; BN bn2;
                const float* wd; BN bnd; };

    int idx = 0;
    auto nx  = [&]() { return (const float*)d_in[idx++]; };
    auto nbn = [&]() { BN o; o.g = nx(); o.b = nx(); o.m = nx(); o.v = nx(); return o; };
    auto ncb = [&]() { CB o; o.w = nx(); o.bn = nbn(); return o; };
    auto nrb = [&](bool hasD) {
        RB o; o.w1 = nx(); o.bn1 = nbn(); o.w2 = nx(); o.bn2 = nbn();
        o.wd = nullptr;
        if (hasD) { o.wd = nx(); o.bnd = nbn(); }
        return o;
    };

    // ---- inputs (setup_inputs dict order, params flattened in insertion order)
    const float* x0    = nx();
    const float* mask0 = nx();
    const float* mask1 = nx();
    const float* mask2 = nx();
    CB stem  = ncb();
    CB s1d   = ncb();
    RB s1r1  = nrb(true);    // 32 -> 64
    RB s1r2  = nrb(false);   // 64 -> 64
    CB s2d   = ncb();
    RB s2r1  = nrb(true);    // 64 -> 128
    RB s2r2  = nrb(false);   // 128 -> 128
    CB up1de = ncb();        // 128 -> 96
    RB up1r1 = nrb(true);    // 160 -> 96
    RB up1r2 = nrb(false);   // 96 -> 96
    CB up2de = ncb();        // 96 -> 96
    RB up2r1 = nrb(true);    // 128 -> 96
    RB up2r2 = nrb(false);   // 96 -> 96

    // ---- workspace: byte bump allocator, 256B aligned ----
    char* wsb = (char*)d_ws;
    size_t off = 0;
    auto alloc = [&](size_t bytes) {
        void* p = wsb + off;
        off += (bytes + 255) & ~(size_t)255;
        return p;
    };
    auto allocH = [&](size_t nf16) { return (_Float16*)alloc(nf16 * 2); };

    const size_t N0 = 64 * 64 * 64, N1 = 32 * 32 * 32, N2 = 16 * 16 * 16;

    // Zero page for out-of-bounds conv taps (unconditional B loads).
    _Float16* Zp = allocH(512);
    zero_f16<<<2, 256, 0, stream>>>(Zp, 512);

    // Activations, [N][C] f16.  Concat buffers hold both channel ranges.
    _Float16* X0h    = allocH(32  * N0);
    _Float16* Rcat0h = allocH(128 * N0);   // ch 0-95: up2_de out, ch 96-127: h0
    _Float16* T0h    = allocH(96  * N0);
    _Float16* E0h    = allocH(96  * N0);
    _Float16* Rcat1h = allocH(160 * N1);   // ch 0-95: up1_de out, ch 96-159: x1
    _Float16* t1h    = allocH(96  * N1);
    _Float16* e1h    = allocH(96  * N1);
    _Float16* c1h    = allocH(96  * N1);
    _Float16* a1h    = allocH(32  * N1);
    _Float16* b1h    = allocH(64  * N1);
    _Float16* a2h    = allocH(64  * N2);
    _Float16* t2h    = allocH(128 * N2);
    _Float16* e2h    = allocH(128 * N2);
    _Float16* b2h    = allocH(128 * N2);
    _Float16* x2h    = allocH(128 * N2);

    // ---- weight packing (one tiny kernel per tensor, A-fragment order) ----
    auto pack = [&](const float* W, int Cout, int Cin, int ks3) {
        _Float16* Wp = allocH((size_t)Cout * Cin * ks3);
        const int total = (Cout / 16) * ks3 * (Cin / 32) * 32;
        pack_w_f16<<<(total + 255) / 256, 256, 0, stream>>>(W, Wp, Cout, Cin, ks3);
        return Wp;
    };
    _Float16* wpStem   = pack(stem.w,   32, 32, 27);
    _Float16* wpS1d    = pack(s1d.w,    32, 32, 8);
    _Float16* wpS1r1a  = pack(s1r1.w1,  64, 32, 27);
    _Float16* wpS1r1b  = pack(s1r1.w2,  64, 64, 27);
    _Float16* wpS1r1d  = pack(s1r1.wd,  64, 32, 1);
    _Float16* wpS1r2a  = pack(s1r2.w1,  64, 64, 27);
    _Float16* wpS1r2b  = pack(s1r2.w2,  64, 64, 27);
    _Float16* wpS2d    = pack(s2d.w,    64, 64, 8);
    _Float16* wpS2r1a  = pack(s2r1.w1, 128, 64, 27);
    _Float16* wpS2r1b  = pack(s2r1.w2, 128, 128, 27);
    _Float16* wpS2r1d  = pack(s2r1.wd, 128, 64, 1);
    _Float16* wpS2r2a  = pack(s2r2.w1, 128, 128, 27);
    _Float16* wpS2r2b  = pack(s2r2.w2, 128, 128, 27);
    _Float16* wpU1de   = pack(up1de.w,  96, 128, 8);
    _Float16* wpU1r1a  = pack(up1r1.w1, 96, 160, 27);
    _Float16* wpU1r1b  = pack(up1r1.w2, 96, 96, 27);
    _Float16* wpU1r1d  = pack(up1r1.wd, 96, 160, 1);
    _Float16* wpU1r2a  = pack(up1r2.w1, 96, 96, 27);
    _Float16* wpU1r2b  = pack(up1r2.w2, 96, 96, 27);
    _Float16* wpU2de   = pack(up2de.w,  96, 96, 8);
    _Float16* wpU2r1a  = pack(up2r1.w1, 96, 128, 27);
    _Float16* wpU2r1b  = pack(up2r1.w2, 96, 96, 27);
    _Float16* wpU2r1d  = pack(up2r1.wd, 96, 128, 1);
    _Float16* wpU2r2a  = pack(up2r2.w1, 96, 96, 27);
    _Float16* wpU2r2b  = pack(up2r2.w2, 96, 96, 27);

    // ---- input layout transform: [32][N0] f32 -> [N0][32] f16 ----
    {
        const int total = 32 * (int)N0;
        cn_to_nc_f16<<<(total + 255) / 256, 256, 0, stream>>>(x0, X0h, 32, (int)N0);
    }

    auto conv = [&](const _Float16* X, int XCs, int XcOff, int Cin, int Din,
                    const _Float16* Wp, BN bn,
                    const _Float16* add, int ACs, const float* msk, int relu,
                    _Float16* Y, int YCs, int YcOff, float* Y32,
                    int Cout, int Dout, int ks, int stride, int pad) {
        const int Nout = Dout * Dout * Dout;
        const int tiles = (Cout / 32) * (Nout / 32);   // 2x2 blocked
        const int blocks = (tiles + WAVES_PER_BLOCK - 1) / WAVES_PER_BLOCK;
        conv3d_nc_wmma<<<blocks, BLOCK_THREADS, 0, stream>>>(
            X, XCs, XcOff, Wp, Zp, bn.g, bn.b, bn.m, bn.v, add, ACs, msk,
            Y, YCs, YcOff, Y32, Cin, Cout, Din, Dout, ks, stride, pad, relu);
    };
    auto deconv = [&](const _Float16* X, int Cin, int Dc, const _Float16* Wp, BN bn,
                      const float* msk, _Float16* Y, int YCs, int YcOff, int Cout) {
        const int Nc = Dc * Dc * Dc;
        const int tiles = (Cout / 16) * (Nc / 16) * 8;
        const int blocks = (tiles + WAVES_PER_BLOCK - 1) / WAVES_PER_BLOCK;
        deconv3d_nc_wmma<<<blocks, BLOCK_THREADS, 0, stream>>>(
            X, Wp, bn.g, bn.b, bn.m, bn.v, msk, Y, YCs, YcOff, Cin, Cout, Dc);
    };

    // ---- network ----
    conv(X0h, 32, 0, 32, 64, wpStem, stem.bn, nullptr, 0, mask0, 1,
         Rcat0h, 128, 96, nullptr, 32, 64, 3, 1, 1);                       // stem -> h0 (in cat0)
    conv(Rcat0h, 128, 96, 32, 64, wpS1d, s1d.bn, nullptr, 0, mask1, 1,
         a1h, 32, 0, nullptr, 32, 32, 2, 2, 0);                            // s1_down

    conv(a1h, 32, 0, 32, 32, wpS1r1a, s1r1.bn1, nullptr, 0, mask1, 1,
         t1h, 64, 0, nullptr, 64, 32, 3, 1, 1);                            // s1_r1
    conv(a1h, 32, 0, 32, 32, wpS1r1d, s1r1.bnd, nullptr, 0, nullptr, 0,
         e1h, 64, 0, nullptr, 64, 32, 1, 1, 0);
    conv(t1h, 64, 0, 64, 32, wpS1r1b, s1r1.bn2, e1h, 64, mask1, 1,
         b1h, 64, 0, nullptr, 64, 32, 3, 1, 1);

    conv(b1h, 64, 0, 64, 32, wpS1r2a, s1r2.bn1, nullptr, 0, mask1, 1,
         t1h, 64, 0, nullptr, 64, 32, 3, 1, 1);                            // s1_r2 -> x1 (in cat1)
    conv(t1h, 64, 0, 64, 32, wpS1r2b, s1r2.bn2, b1h, 64, mask1, 1,
         Rcat1h, 160, 96, nullptr, 64, 32, 3, 1, 1);

    conv(Rcat1h, 160, 96, 64, 32, wpS2d, s2d.bn, nullptr, 0, mask2, 1,
         a2h, 64, 0, nullptr, 64, 16, 2, 2, 0);                            // s2_down

    conv(a2h, 64, 0, 64, 16, wpS2r1a, s2r1.bn1, nullptr, 0, mask2, 1,
         t2h, 128, 0, nullptr, 128, 16, 3, 1, 1);                          // s2_r1
    conv(a2h, 64, 0, 64, 16, wpS2r1d, s2r1.bnd, nullptr, 0, nullptr, 0,
         e2h, 128, 0, nullptr, 128, 16, 1, 1, 0);
    conv(t2h, 128, 0, 128, 16, wpS2r1b, s2r1.bn2, e2h, 128, mask2, 1,
         b2h, 128, 0, nullptr, 128, 16, 3, 1, 1);

    conv(b2h, 128, 0, 128, 16, wpS2r2a, s2r2.bn1, nullptr, 0, mask2, 1,
         t2h, 128, 0, nullptr, 128, 16, 3, 1, 1);                          // s2_r2 -> x2
    conv(t2h, 128, 0, 128, 16, wpS2r2b, s2r2.bn2, b2h, 128, mask2, 1,
         x2h, 128, 0, nullptr, 128, 16, 3, 1, 1);

    deconv(x2h, 128, 16, wpU1de, up1de.bn, mask1, Rcat1h, 160, 0, 96);     // up1_de (cat with x1)

    conv(Rcat1h, 160, 0, 160, 32, wpU1r1a, up1r1.bn1, nullptr, 0, mask1, 1,
         t1h, 96, 0, nullptr, 96, 32, 3, 1, 1);                            // up1_r1
    conv(Rcat1h, 160, 0, 160, 32, wpU1r1d, up1r1.bnd, nullptr, 0, nullptr, 0,
         e1h, 96, 0, nullptr, 96, 32, 1, 1, 0);
    conv(t1h, 96, 0, 96, 32, wpU1r1b, up1r1.bn2, e1h, 96, mask1, 1,
         c1h, 96, 0, nullptr, 96, 32, 3, 1, 1);

    conv(c1h, 96, 0, 96, 32, wpU1r2a, up1r2.bn1, nullptr, 0, mask1, 1,
         t1h, 96, 0, nullptr, 96, 32, 3, 1, 1);                            // up1_r2 -> e1h
    conv(t1h, 96, 0, 96, 32, wpU1r2b, up1r2.bn2, c1h, 96, mask1, 1,
         e1h, 96, 0, nullptr, 96, 32, 3, 1, 1);

    deconv(e1h, 96, 32, wpU2de, up2de.bn, mask0, Rcat0h, 128, 0, 96);      // up2_de (cat with h0)

    conv(Rcat0h, 128, 0, 128, 64, wpU2r1a, up2r1.bn1, nullptr, 0, mask0, 1,
         T0h, 96, 0, nullptr, 96, 64, 3, 1, 1);                            // up2_r1
    conv(Rcat0h, 128, 0, 128, 64, wpU2r1d, up2r1.bnd, nullptr, 0, nullptr, 0,
         E0h, 96, 0, nullptr, 96, 64, 1, 1, 0);
    conv(T0h, 96, 0, 96, 64, wpU2r1b, up2r1.bn2, E0h, 96, mask0, 1,
         E0h, 96, 0, nullptr, 96, 64, 3, 1, 1);                            // in-place add (elementwise-safe)

    conv(E0h, 96, 0, 96, 64, wpU2r2a, up2r2.bn1, nullptr, 0, mask0, 1,
         T0h, 96, 0, nullptr, 96, 64, 3, 1, 1);                            // up2_r2 -> d_out (f32 [C][N])
    conv(T0h, 96, 0, 96, 64, wpU2r2b, up2r2.bn2, E0h, 96, mask0, 1,
         nullptr, 0, 0, (float*)d_out, 96, 64, 3, 1, 1);
}
